// DependencyEncoder_41291815584282
// MI455X (gfx1250) — matrix operand order repaired and checked by reading, not compiled, not run on hardware
//
#include <hip/hip_runtime.h>

#define Bsz    512
#define Lsz    64
#define Hsz    256
#define TDsz   64
#define Tsteps 127
#define MAXS   66
#define ROWS   16
#define NBLK   (Bsz / ROWS)   // 32 workgroups

typedef __attribute__((ext_vector_type(16))) __bf16 v16bf;
typedef __attribute__((ext_vector_type(8)))  float  v8f;

// ---- bf16 weight workspace layout (element offsets) ----
#define SZ_WIH (4 * TDsz * 3 * Hsz)   // 256 x 768
#define SZ_WHH (4 * TDsz * TDsz)      // 256 x 64
#define SZ_UL  (5 * Hsz * Hsz)        // 1280 x 256
#define SZ_UR  (5 * Hsz * Hsz)        // 1280 x 256
#define SZ_WX  (5 * Hsz * TDsz)       // 1280 x 64
#define OFF_WIH 0
#define OFF_WHH (OFF_WIH + SZ_WIH)
#define OFF_UL  (OFF_WHH + SZ_WHH)
#define OFF_UR  (OFF_UL + SZ_UL)
#define OFF_WX  (OFF_UR + SZ_UR)
#define NW_TOT  (OFF_WX + SZ_WX)      // 950272 bf16 elements
#define STACK_BYTE_OFF ((size_t)NW_TOT * 2)          // 1900544, 256B aligned
#define STACK_ELEMS    ((size_t)Bsz * MAXS * Hsz)    // per Sh / Sc

__device__ __forceinline__ float sigf(float x) { return 1.0f / (1.0f + expf(-x)); }

// A-fragment: 16xK row-major bf16 in LDS (padded stride). A layout:
// lanes 0-15: M=lane, K = k0+[0..15]; lanes 16-31: M=lane-16, K = k0+[16..31].
__device__ __forceinline__ v16bf load_a(const __bf16* lds, int strideK, int k0, int lane) {
  const __bf16* p = lds + (lane & 15) * strideK + k0 + 16 * (lane >> 4);
  v16bf a;
#pragma unroll
  for (int j = 0; j < 16; ++j) a[j] = p[j];
  return a;
}

// B-fragment: W is [N][K] row-major bf16 in global; B = W^T tile (k0,n0).
// lanes 0-15: N=lane, K=k0+0..15; lanes 16-31: N=lane-16, K=k0+16..31 (32B contiguous).
__device__ __forceinline__ v16bf load_b(const __bf16* __restrict__ W, int K, int n0, int k0, int lane) {
  const __bf16* p = W + (long)(n0 + (lane & 15)) * K + k0 + 16 * (lane >> 4);
  v16bf b;
#pragma unroll
  for (int j = 0; j < 16; ++j) b[j] = p[j];
  return b;
}

#define WMMA_BF16(A, B, C) \
  __builtin_amdgcn_wmma_f32_16x16x32_bf16(false, (A), false, (B), (short)0, (C), false, false)

__global__ void convert_weights_kernel(const float* __restrict__ W_x,
                                       const float* __restrict__ U_r,
                                       const float* __restrict__ U_l,
                                       const float* __restrict__ W_ih,
                                       const float* __restrict__ W_hh,
                                       __bf16* __restrict__ wsb) {
  for (long i = (long)blockIdx.x * blockDim.x + threadIdx.x; i < (long)NW_TOT;
       i += (long)gridDim.x * blockDim.x) {
    float v;
    if (i < OFF_WHH)      v = W_ih[i - OFF_WIH];
    else if (i < OFF_UL)  v = W_hh[i - OFF_WHH];
    else if (i < OFF_UR)  v = U_l[i - OFF_UL];
    else if (i < OFF_WX)  v = U_r[i - OFF_UR];
    else                  v = W_x[i - OFF_WX];
    wsb[i] = (__bf16)v;
  }
}

__global__ __launch_bounds__(256)
void spinn_persistent_kernel(const float* __restrict__ tokens_h,
                             const float* __restrict__ tokens_c,
                             const int*   __restrict__ transitions,
                             const float* __restrict__ b_l,
                             const float* __restrict__ b_ih,
                             const float* __restrict__ b_hh,
                             const float* __restrict__ th0,
                             const float* __restrict__ tc0,
                             const __bf16* __restrict__ wsb,
                             float* __restrict__ Sh,
                             float* __restrict__ Sc,
                             float* __restrict__ out) {
  __shared__ __bf16 xA[ROWS][3 * Hsz + 16];   // tracking input [buf|top|sec], bf16
  __shared__ __bf16 thA[ROWS][TDsz + 16];     // tracking hidden state, bf16 (A for W_hh/W_x)
  __shared__ __bf16 chilA[ROWS][Hsz + 16];
  __shared__ __bf16 headA[ROWS][Hsz + 16];
  __shared__ float  chilC[ROWS][Hsz + 8];
  __shared__ float  headC[ROWS][Hsz + 8];
  __shared__ float  hbuf[ROWS][Hsz + 8];
  __shared__ float  cbuf[ROWS][Hsz + 8];
  __shared__ float  tcS[ROWS][TDsz + 8];      // tracking cell state f32
  __shared__ int s_p[ROWS], s_bp[ROWS], s_tr[ROWS];
  __shared__ int s_anyred;

  const int tid  = threadIdx.x;
  const int lane = tid & 31;
  const int wv   = tid >> 5;          // 8 waves
  const int m    = tid >> 4;          // batch row 0..15 (16 threads per row)
  const int cb   = (tid & 15) * 16;   // 16-col slice of H
  const long b   = (long)blockIdx.x * ROWS + m;

  const __bf16* Wih = wsb + OFF_WIH;
  const __bf16* Whh = wsb + OFF_WHH;
  const __bf16* Ul  = wsb + OFF_UL;
  const __bf16* Ur  = wsb + OFF_UR;
  const __bf16* Wx  = wsb + OFF_WX;

  // -------- init: stack slots 0,1 = tokens[0]; tracking state from th0/tc0 --------
  {
    const float* th_s = tokens_h + b * Lsz * Hsz;
    const float* tc_s = tokens_c + b * Lsz * Hsz;
    float* sh0 = Sh + b * MAXS * Hsz;
    float* sc0 = Sc + b * MAXS * Hsz;
#pragma unroll 4
    for (int j = 0; j < 16; ++j) {
      float hv = th_s[cb + j], cv = tc_s[cb + j];
      sh0[cb + j] = hv; sh0[Hsz + cb + j] = hv;
      sc0[cb + j] = cv; sc0[Hsz + cb + j] = cv;
    }
    int n0 = (tid & 15) * 4;  // 4 of the 64 TD columns
#pragma unroll
    for (int j = 0; j < 4; ++j) {
      tcS[m][n0 + j] = tc0[b * TDsz + n0 + j];
      thA[m][n0 + j] = (__bf16)th0[b * TDsz + n0 + j];
    }
    if (tid < ROWS) { s_p[tid] = 2; s_bp[tid] = 0; }
  }
  __syncthreads();

  for (int t = 0; t < Tsteps; ++t) {
    if (tid == 0) s_anyred = 0;
    __syncthreads();
    if (tid < ROWS) {
      int tr = transitions[((long)blockIdx.x * ROWS + tid) * Tsteps + t];
      s_tr[tid] = tr;
      if (tr == 2 || tr == 3) s_anyred = 1;
    }
    __syncthreads();
    const int anyred = s_anyred;

    // -------- load/stage phase --------
    {
      int p = s_p[m], bp = s_bp[m];
      bool m2 = (s_tr[m] == 2);
      const float* bufh = tokens_h + (b * Lsz + bp) * Hsz;
      const float* toph = Sh + (b * MAXS + p - 1) * Hsz;
      const float* sech = Sh + (b * MAXS + p - 2) * Hsz;
      const float* topc = Sc + (b * MAXS + p - 1) * Hsz;
      const float* secc = Sc + (b * MAXS + p - 2) * Hsz;
#pragma unroll 4
      for (int j = 0; j < 16; ++j) {
        int c = cb + j;
        float bh = bufh[c], thv = toph[c], shv = sech[c];
        xA[m][c]           = (__bf16)bh;
        xA[m][Hsz + c]     = (__bf16)thv;
        xA[m][2 * Hsz + c] = (__bf16)shv;
        if (anyred) {
          float tcv = topc[c], scv = secc[c];
          headA[m][c] = (__bf16)(m2 ? thv : shv);
          chilA[m][c] = (__bf16)(m2 ? shv : thv);
          headC[m][c] = m2 ? tcv : scv;
          chilC[m][c] = m2 ? scv : tcv;
        }
      }
    }
    __syncthreads();

    // -------- tracking LSTM GEMM: g[16,256] = x@W_ih^T + th@W_hh^T + biases ----
    // waves 0..3; wave w owns gate tiles {w, w+4, w+8, w+12} -> all 4 gates of its
    // 16-column slice of TD, kept in registers.
    v8f tacc[4];
    if (wv < 4) {
#pragma unroll
      for (int j = 0; j < 4; ++j) {
        int col = 16 * wv + 64 * j + (lane & 15);
        float bias = b_ih[col] + b_hh[col];
#pragma unroll
        for (int r = 0; r < 8; ++r) tacc[j][r] = bias;
      }
      for (int k0 = 0; k0 < 3 * Hsz; k0 += 32) {
        v16bf a = load_a(&xA[0][0], 3 * Hsz + 16, k0, lane);
#pragma unroll
        for (int j = 0; j < 4; ++j) {
          v16bf bm = load_b(Wih, 3 * Hsz, 16 * wv + 64 * j, k0, lane);
          tacc[j] = WMMA_BF16(a, bm, tacc[j]);
        }
      }
#pragma unroll
      for (int k0 = 0; k0 < TDsz; k0 += 32) {
        v16bf a = load_a(&thA[0][0], TDsz + 16, k0, lane);
#pragma unroll
        for (int j = 0; j < 4; ++j) {
          v16bf bm = load_b(Whh, TDsz, 16 * wv + 64 * j, k0, lane);
          tacc[j] = WMMA_BF16(a, bm, tacc[j]);
        }
      }
    }
    __syncthreads();  // all reads of old thA done before it is overwritten

    // -------- tracking LSTM nonlinearity (in-register gates) --------
    if (wv < 4) {
      int hlf = lane >> 4;
      int n = 16 * wv + (lane & 15);
#pragma unroll
      for (int r = 0; r < 8; ++r) {
        int mm = r + 8 * hlf;
        float gi = tacc[0][r], gf = tacc[1][r], gg = tacc[2][r], go = tacc[3][r];
        float tcn = sigf(gf) * tcS[mm][n] + sigf(gi) * tanhf(gg);
        float thn = sigf(go) * tanhf(tcn);
        tcS[mm][n] = tcn;
        thA[mm][n] = (__bf16)thn;
      }
    }
    __syncthreads();

    // -------- TreeLSTM composition (skipped when every row shifts) --------
    if (anyred) {
      // 80 gate tiles (5*H/16); wave w owns tiles w+8j -> all 5 gates for output
      // column groups g=w and g=w+8 live in this wave's registers.
      v8f acc[10];
#pragma unroll
      for (int j = 0; j < 10; ++j) {
        float bias = b_l[16 * (wv + 8 * j) + (lane & 15)];
#pragma unroll
        for (int r = 0; r < 8; ++r) acc[j][r] = bias;
      }
      for (int k0 = 0; k0 < Hsz; k0 += 32) {
        v16bf a = load_a(&chilA[0][0], Hsz + 16, k0, lane);
#pragma unroll
        for (int j = 0; j < 10; ++j) {
          v16bf bm = load_b(Ul, Hsz, 16 * (wv + 8 * j), k0, lane);
          acc[j] = WMMA_BF16(a, bm, acc[j]);
        }
      }
      for (int k0 = 0; k0 < Hsz; k0 += 32) {
        v16bf a = load_a(&headA[0][0], Hsz + 16, k0, lane);
#pragma unroll
        for (int j = 0; j < 10; ++j) {
          v16bf bm = load_b(Ur, Hsz, 16 * (wv + 8 * j), k0, lane);
          acc[j] = WMMA_BF16(a, bm, acc[j]);
        }
      }
#pragma unroll
      for (int k0 = 0; k0 < TDsz; k0 += 32) {
        v16bf a = load_a(&thA[0][0], TDsz + 16, k0, lane);
#pragma unroll
        for (int j = 0; j < 10; ++j) {
          v16bf bm = load_b(Wx, TDsz, 16 * (wv + 8 * j), k0, lane);
          acc[j] = WMMA_BF16(a, bm, acc[j]);
        }
      }
      // gates order: i | o | f_l | f_r | u  -> acc[gsel], acc[2+gsel], acc[4+gsel],
      // acc[6+gsel], acc[8+gsel] for output group g = wv + 8*gsel.
      int hlf = lane >> 4;
#pragma unroll
      for (int gsel = 0; gsel < 2; ++gsel) {
        int n = 16 * (wv + 8 * gsel) + (lane & 15);
#pragma unroll
        for (int r = 0; r < 8; ++r) {
          int mm = r + 8 * hlf;
          float cc = sigf(acc[0 + gsel][r]) * tanhf(acc[8 + gsel][r])
                   + sigf(acc[4 + gsel][r]) * chilC[mm][n]
                   + sigf(acc[6 + gsel][r]) * headC[mm][n];
          float hh = sigf(acc[2 + gsel][r]) * tanhf(cc);
          cbuf[mm][n] = cc;
          hbuf[mm][n] = hh;
        }
      }
      __syncthreads();
    }

    // -------- stack writeback (each stack cell touched by exactly one thread) ----
    {
      int tr = s_tr[m];
      int p = s_p[m], bp = s_bp[m];
      if (tr == 1) {  // shift: push buffer token at slot p
        const float* bh = tokens_h + (b * Lsz + bp) * Hsz;
        const float* bc = tokens_c + (b * Lsz + bp) * Hsz;
        float* dh = Sh + (b * MAXS + p) * Hsz;
        float* dc = Sc + (b * MAXS + p) * Hsz;
#pragma unroll 4
        for (int j = 0; j < 16; ++j) { dh[cb + j] = bh[cb + j]; dc[cb + j] = bc[cb + j]; }
      } else if (tr == 2 || tr == 3) {  // reduce: write composed node at slot p-2
        float* dh = Sh + (b * MAXS + p - 2) * Hsz;
        float* dc = Sc + (b * MAXS + p - 2) * Hsz;
#pragma unroll 4
        for (int j = 0; j < 16; ++j) { dh[cb + j] = hbuf[m][cb + j]; dc[cb + j] = cbuf[m][cb + j]; }
      }  // else: reference rewrites top at p-1 (no-op), p unchanged
    }
    __syncthreads();
    if (tid < ROWS) {
      int tr = s_tr[tid];
      if (tr == 1) { s_p[tid] += 1; s_bp[tid] += 1; }
      else if (tr == 2 || tr == 3) { s_p[tid] -= 1; }
    }
  }
  __syncthreads();

  // -------- output: Sh[b, p-1] --------
  {
    int p = s_p[m];
    const float* src = Sh + (b * MAXS + p - 1) * Hsz;
#pragma unroll 4
    for (int j = 0; j < 16; ++j) out[b * Hsz + cb + j] = src[cb + j];
  }
}

extern "C" void kernel_launch(void* const* d_in, const int* in_sizes, int n_in,
                              void* d_out, int out_size, void* d_ws, size_t ws_size,
                              hipStream_t stream) {
  const float* tokens_h    = (const float*)d_in[0];
  const float* tokens_c    = (const float*)d_in[1];
  const int*   transitions = (const int*)d_in[2];
  const float* W_x  = (const float*)d_in[3];
  const float* U_r  = (const float*)d_in[4];
  const float* U_l  = (const float*)d_in[5];
  const float* b_l  = (const float*)d_in[6];
  const float* W_ih = (const float*)d_in[7];
  const float* W_hh = (const float*)d_in[8];
  const float* b_ih = (const float*)d_in[9];
  const float* b_hh = (const float*)d_in[10];
  const float* th0  = (const float*)d_in[11];
  const float* tc0  = (const float*)d_in[12];

  __bf16* wsb = (__bf16*)d_ws;
  float* Sh = (float*)((char*)d_ws + STACK_BYTE_OFF);
  float* Sc = Sh + STACK_ELEMS;
  float* out = (float*)d_out;

  convert_weights_kernel<<<(NW_TOT + 255) / 256, 256, 0, stream>>>(W_x, U_r, U_l, W_ih, W_hh, wsb);
  spinn_persistent_kernel<<<NBLK, 256, 0, stream>>>(tokens_h, tokens_c, transitions,
                                                    b_l, b_ih, b_hh, th0, tc0,
                                                    wsb, Sh, Sc, out);
}
